// EdgeTokenTransformer_61083024883714
// MI455X (gfx1250) — matrix operand order repaired and checked by reading, not compiled, not run on hardware
//
#include <hip/hip_runtime.h>

typedef __attribute__((ext_vector_type(16))) _Float16 v16h;
typedef __attribute__((ext_vector_type(8)))  _Float16 v8h;
typedef __attribute__((ext_vector_type(4)))  _Float16 v4h;
typedef __attribute__((ext_vector_type(8)))  float    v8f;
typedef __attribute__((ext_vector_type(4)))  int      v4i;

constexpr int Hh  = 256;
constexpr int Ll  = 3;
constexpr int NHh = 8;
constexpr int DHh = 32;
constexpr int EHh = 512;
constexpr int TPc = 16;
constexpr int Bb  = 8;
constexpr int Vv  = 64;
constexpr int Ec  = Vv * (Vv - 1) / 2;   // 2016
constexpr int Mrows = Bb * Ec;           // 16128
static_assert(Ec % 32 == 0, "E must be multiple of 32");
static_assert(Mrows % 128 == 0, "M must be multiple of 128");

// ---------------------------------------------------------------------------
// gfx1250 async global->LDS copy (ASYNCcnt-tracked), with safe fallback.
#if defined(__has_builtin)
#  if __has_builtin(__builtin_amdgcn_global_load_async_to_lds_b128) && \
      __has_builtin(__builtin_amdgcn_s_wait_asynccnt)
#    define USE_ASYNC_LDS 1
#  endif
#endif
#ifndef USE_ASYNC_LDS
#  define USE_ASYNC_LDS 0
#endif

typedef __attribute__((address_space(1))) v4i gas_v4i;   // global int4
typedef __attribute__((address_space(3))) v4i las_v4i;   // LDS int4

__device__ __forceinline__ void copy16_g2lds(const _Float16* g, _Float16* l) {
#if USE_ASYNC_LDS
    __builtin_amdgcn_global_load_async_to_lds_b128((gas_v4i*)g, (las_v4i*)l, 0, 0);
#else
    *(v8h*)l = *(const v8h*)g;
#endif
}
__device__ __forceinline__ void async_fence() {
#if USE_ASYNC_LDS
    __builtin_amdgcn_s_wait_asynccnt(0);
#endif
}

__device__ __forceinline__ float gelu_f(float x) {
    return 0.5f * x * (1.0f + erff(x * 0.70710678118654752f));
}

// ---------------------------------------------------------------------------
__global__ void fill_zero_kernel(float* p, int n) {
    int i = blockIdx.x * blockDim.x + threadIdx.x;
    if (i < n) p[i] = 0.0f;
}

// weight convert f32 -> f16, layout preserved ([out(N)][in(K)] row-major)
__global__ void wt_convert_kernel(const float* __restrict__ src, _Float16* __restrict__ dst, int n) {
    int i = blockIdx.x * blockDim.x + threadIdx.x;
    if (i < n) dst[i] = (_Float16)src[i];
}

// ---------------------------------------------------------------------------
// time embedding MLP (tiny)
__global__ void time_hidden_kernel(const float* __restrict__ t, const float* __restrict__ w1,
                                   const float* __restrict__ b1, float* __restrict__ thid) {
    int idx = blockIdx.x * blockDim.x + threadIdx.x;  // B*256
    int b = idx >> 8, h = idx & 255;
    float tb = t[b];
    float acc = b1[h];
    #pragma unroll
    for (int k = 0; k < TPc; ++k) {
        float fr = __expf(-9.210340371976184f * (float)k / (float)TPc);
        float ang = tb * fr;
        acc += __sinf(ang) * w1[h * 2 * TPc + k] + __cosf(ang) * w1[h * 2 * TPc + TPc + k];
    }
    thid[idx] = gelu_f(acc);
}

__global__ void time_vec_kernel(const float* __restrict__ thid, const float* __restrict__ w2,
                                const float* __restrict__ b2, float* __restrict__ tvec) {
    int idx = blockIdx.x * blockDim.x + threadIdx.x;  // B*256
    int b = idx >> 8, h = idx & 255;
    float acc = b2[h];
    for (int q = 0; q < Hh; ++q) acc += thid[b * Hh + q] * w2[h * Hh + q];
    tvec[idx] = acc;
}

// ---------------------------------------------------------------------------
__global__ void deg_kernel(const float* __restrict__ adj, float* __restrict__ dnorm) {
    __shared__ float dsh[Vv];
    int b = blockIdx.x;
    int v = threadIdx.x;  // 64 threads
    float s = 0.f;
    for (int u = 0; u < Vv; ++u) s += adj[((size_t)b * Vv + v) * Vv + u];
    dsh[v] = s;
    __syncthreads();
    float mx = 1.0f;
    for (int u = 0; u < Vv; ++u) mx = fmaxf(mx, dsh[u]);
    dnorm[b * Vv + v] = s / mx;
}

// ---------------------------------------------------------------------------
__global__ __launch_bounds__(256)
void embed_kernel(const int* __restrict__ obs, const float* __restrict__ tok_emb,
                  const float* __restrict__ iemb, const float* __restrict__ jemb,
                  const float* __restrict__ dnorm, const float* __restrict__ tvec,
                  const float* __restrict__ dw1, const float* __restrict__ db1,
                  const float* __restrict__ dw2, const float* __restrict__ db2,
                  float* __restrict__ xf32, _Float16* __restrict__ xf16) {
    __shared__ float hid[Hh];
    int be = blockIdx.x;
    int b = be / Ec, e = be % Ec;
    int i = 0, rem = e;
    while (rem >= Vv - 1 - i) { rem -= Vv - 1 - i; ++i; }
    int j = i + 1 + rem;
    int h = threadIdx.x;
    float di = dnorm[b * Vv + i], dj = dnorm[b * Vv + j];
    hid[h] = gelu_f(dw1[h * 2 + 0] * di + dw1[h * 2 + 1] * dj + db1[h]);
    __syncthreads();
    int tok = obs[((size_t)b * Vv + i) * Vv + j];
    tok = min(max(tok, 0), 3);
    float acc = tok_emb[tok * Hh + h] + iemb[i * Hh + h] + jemb[j * Hh + h] +
                tvec[b * Hh + h] + db2[h];
    for (int q = 0; q < Hh; ++q) acc += hid[q] * dw2[h * Hh + q];
    xf32[(size_t)be * Hh + h] = acc;
    xf16[(size_t)be * Hh + h] = (_Float16)acc;
}

// ---------------------------------------------------------------------------
// f16 WMMA GEMM: C[M,N] = A[M,K] @ Bw^T + bias, Bw stored [N][K] row-major.
// block 256 (8 waves), tile 128x64, per-wave 32x32 (2x2 WMMA register tile),
// K step 32, double-buffered async LDS staging.
// mode 0: outF32 = val (+resid)   mode 1: outF16 = val
// mode 2: QKV scatter into [B,NH,E,DH] f16 (Q pre-scaled by 1/sqrt(DH))
__global__ __launch_bounds__(256)
void gemm_kernel(const _Float16* __restrict__ A, const _Float16* __restrict__ Bw,
                 const float* __restrict__ bias, int M, int N, int K,
                 int mode, int act,
                 float* __restrict__ outF32, const float* __restrict__ resid,
                 _Float16* __restrict__ outF16,
                 _Float16* __restrict__ qdst, _Float16* __restrict__ kdst,
                 _Float16* __restrict__ vdst) {
    __shared__ alignas(64) _Float16 Asm[2][128][32];   // [m][k]
    __shared__ alignas(64) _Float16 Bsm[2][64][32];    // [n][k]
    int tid = threadIdx.x;
    int lane = tid & 31;
    int w = tid >> 5;
    int Mbase = blockIdx.y * 128;
    int Nbase = blockIdx.x * 64;
    int wm = (w >> 1) * 32;   // 4 wave rows
    int wn = (w & 1) * 32;    // 2 wave cols
    int l16 = lane & 15;
    int hf = lane >> 4;

    auto stage = [&](int buf, int kt) {
        // A tile: 128x32 f16 = 512 16B chunks, 2 per thread
        #pragma unroll
        for (int rep = 0; rep < 2; ++rep) {
            int c = tid + rep * 256;
            int r = c >> 2, kc = (c & 3) * 8;
            copy16_g2lds(A + (size_t)(Mbase + r) * K + kt + kc, &Asm[buf][r][kc]);
        }
        // B tile: 64x32 f16 = 256 16B chunks, 1 per thread (straight copy)
        {
            int n = tid >> 2, kc = (tid & 3) * 8;
            copy16_g2lds(Bw + (size_t)(Nbase + n) * K + kt + kc, &Bsm[buf][n][kc]);
        }
    };

    v8f acc00 = {}, acc01 = {}, acc10 = {}, acc11 = {};
    int nkt = K >> 5;
    stage(0, 0);
    for (int it = 0; it < nkt; ++it) {
        async_fence();
        __syncthreads();
        if (it + 1 < nkt) stage((it + 1) & 1, (it + 1) << 5);
        int buf = it & 1;
        // A fragments: lane hf=0 -> K {0..15}; hf=1 -> K {16..31}
        int kb = hf ? 16 : 0;
        v16h a0, a1;
        {
            v8h c0 = *(const v8h*)&Asm[buf][wm + l16][kb];
            v8h c1 = *(const v8h*)&Asm[buf][wm + l16][kb + 8];
            v8h c2 = *(const v8h*)&Asm[buf][wm + 16 + l16][kb];
            v8h c3 = *(const v8h*)&Asm[buf][wm + 16 + l16][kb + 8];
            #pragma unroll
            for (int q = 0; q < 8; ++q) {
                a0[q] = c0[q]; a0[q + 8] = c1[q];
                a1[q] = c2[q]; a1[q + 8] = c3[q];
            }
        }
        // B fragments: lane holds K contiguous (0..15 | 16..31) at fixed n
        int kb2 = hf ? 16 : 0;
        v16h b0 = *(const v16h*)&Bsm[buf][wn + l16][kb2];
        v16h b1 = *(const v16h*)&Bsm[buf][wn + 16 + l16][kb2];
        acc00 = __builtin_amdgcn_wmma_f32_16x16x32_f16(false, a0, false, b0, (short)0, acc00, false, false);
        acc01 = __builtin_amdgcn_wmma_f32_16x16x32_f16(false, a0, false, b1, (short)0, acc01, false, false);
        acc10 = __builtin_amdgcn_wmma_f32_16x16x32_f16(false, a1, false, b0, (short)0, acc10, false, false);
        acc11 = __builtin_amdgcn_wmma_f32_16x16x32_f16(false, a1, false, b1, (short)0, acc11, false, false);
    }

    auto epi = [&](const v8f& acc, int ti, int tj) {
        float bn = bias ? bias[Nbase + wn + tj * 16 + l16] : 0.f;
        #pragma unroll
        for (int v = 0; v < 8; ++v) {
            int m = Mbase + wm + ti * 16 + v + 8 * hf;   // C layout: row = reg + 8*(lane>=16)
            int n = Nbase + wn + tj * 16 + l16;
            float x = acc[v] + bn;
            if (act) x = gelu_f(x);
            if (mode == 0) {
                float r = resid ? resid[(size_t)m * N + n] : 0.f;
                outF32[(size_t)m * N + n] = x + r;
            } else if (mode == 1) {
                outF16[(size_t)m * N + n] = (_Float16)x;
            } else {
                int part = n >> 8;
                int head = (n >> 5) & 7;
                int d = n & 31;
                int bI = m / Ec, e = m - bI * Ec;
                size_t dst = (((size_t)bI * NHh + head) * Ec + e) * DHh + d;
                _Float16* p = (part == 0) ? qdst : (part == 1) ? kdst : vdst;
                float val = (part == 0) ? x * 0.17677669529663687f : x;
                p[dst] = (_Float16)val;
            }
        }
    };
    epi(acc00, 0, 0); epi(acc01, 0, 1); epi(acc10, 1, 0); epi(acc11, 1, 1);
}

// ---------------------------------------------------------------------------
// Flash attention: grid (E/128, B*NH), block 256 (8 waves x 16 q-rows),
// double-buffered K/V tiles (K async-staged), online softmax in 16-lane halves.
__global__ __launch_bounds__(256)
void attn_kernel(const _Float16* __restrict__ Qh, const _Float16* __restrict__ Kh,
                 const _Float16* __restrict__ Vh, _Float16* __restrict__ attnOut) {
    __shared__ alignas(64) _Float16 Ks[2][32][32];     // [key][d]
    __shared__ alignas(64) _Float16 Vt[2][32][32];     // [d][key] (transposed)
    __shared__ alignas(64) _Float16 Ps[8][16][32];     // per-wave P staging [row][key]
    int tid = threadIdx.x, lane = tid & 31, w = tid >> 5;
    int bh = blockIdx.y;
    int b = bh >> 3, h = bh & 7;
    int qbase = blockIdx.x * 128 + w * 16;
    int l16 = lane & 15, hf = lane >> 4;
    const _Float16* Qp = Qh + (size_t)bh * Ec * DHh;
    const _Float16* Kp = Kh + (size_t)bh * Ec * DHh;
    const _Float16* Vp = Vh + (size_t)bh * Ec * DHh;

    auto stage = [&](int buf, int kt) {
        if (tid < 128) {   // K tile: 128 16B chunks, async
            int key = tid >> 2, kc = (tid & 3) * 8;
            copy16_g2lds(Kp + (size_t)(kt + key) * DHh + kc, &Ks[buf][key][kc]);
        }
        {   // V tile transposed (manual)
            int key = tid >> 3, dc = (tid & 7) * 4;
            v4h vd = *(const v4h*)(Vp + (size_t)(kt + key) * DHh + dc);
            #pragma unroll
            for (int q = 0; q < 4; ++q) Vt[buf][dc + q][key] = vd[q];
        }
    };

    // Q fragment (held for whole kernel)
    v16h aQ;
    {
        int q = qbase + l16;
        if (q >= Ec) q = Ec - 1;
        int db = hf ? 16 : 0;
        const _Float16* p = Qp + (size_t)q * DHh;
        v8h c0 = *(const v8h*)(p + db);
        v8h c1 = *(const v8h*)(p + db + 8);
        #pragma unroll
        for (int i = 0; i < 8; ++i) { aQ[i] = c0[i]; aQ[i + 8] = c1[i]; }
    }

    v8f o0 = {}, o1 = {};
    float mrow[8], lrow[8];
    #pragma unroll
    for (int v = 0; v < 8; ++v) { mrow[v] = -1e30f; lrow[v] = 0.f; }

    constexpr int NT = Ec / 32;   // 63
    stage(0, 0);
    for (int it = 0; it < NT; ++it) {
        async_fence();
        __syncthreads();
        if (it + 1 < NT) stage((it + 1) & 1, (it + 1) * 32);
        int buf = it & 1;

        // S = Q @ K^T : two 16x16 tiles (keys 0..15 and 16..31), K-dim = DH = 32
        int kb = hf ? 16 : 0;
        v16h bK0 = *(const v16h*)&Ks[buf][l16][kb];
        v16h bK1 = *(const v16h*)&Ks[buf][16 + l16][kb];
        v8f z = {};
        v8f s0 = __builtin_amdgcn_wmma_f32_16x16x32_f16(false, aQ, false, bK0, (short)0, z, false, false);
        v8f s1 = __builtin_amdgcn_wmma_f32_16x16x32_f16(false, aQ, false, bK1, (short)0, z, false, false);

        // online softmax (row stats live in 16-lane halves of the C layout)
        #pragma unroll
        for (int v = 0; v < 8; ++v) {
            float tmax = fmaxf(s0[v], s1[v]);
            #pragma unroll
            for (int msk = 8; msk >= 1; msk >>= 1) tmax = fmaxf(tmax, __shfl_xor(tmax, msk, 32));
            float mnew = fmaxf(mrow[v], tmax);
            float sc = __expf(mrow[v] - mnew);
            float p0 = __expf(s0[v] - mnew);
            float p1 = __expf(s1[v] - mnew);
            float rs = p0 + p1;
            #pragma unroll
            for (int msk = 8; msk >= 1; msk >>= 1) rs += __shfl_xor(rs, msk, 32);
            lrow[v] = lrow[v] * sc + rs;
            mrow[v] = mnew;
            o0[v] *= sc;
            o1[v] *= sc;
            int r = v + 8 * hf;
            Ps[w][r][l16] = (_Float16)p0;
            Ps[w][r][16 + l16] = (_Float16)p1;
        }

        // P as A-fragment (wave-private LDS round trip does the lane shuffle)
        v16h aP;
        {
            int kb2 = hf ? 16 : 0;
            v8h c0 = *(const v8h*)&Ps[w][l16][kb2];
            v8h c1 = *(const v8h*)&Ps[w][l16][kb2 + 8];
            #pragma unroll
            for (int i = 0; i < 8; ++i) { aP[i] = c0[i]; aP[i + 8] = c1[i]; }
        }
        // V fragments from transposed tile: contiguous 16 keys at fixed d
        int koff = hf ? 16 : 0;
        v16h bV0 = *(const v16h*)&Vt[buf][l16][koff];
        v16h bV1 = *(const v16h*)&Vt[buf][16 + l16][koff];
        o0 = __builtin_amdgcn_wmma_f32_16x16x32_f16(false, aP, false, bV0, (short)0, o0, false, false);
        o1 = __builtin_amdgcn_wmma_f32_16x16x32_f16(false, aP, false, bV1, (short)0, o1, false, false);
    }

    #pragma unroll
    for (int v = 0; v < 8; ++v) {
        int q = qbase + v + 8 * hf;
        if (q < Ec) {
            float inv = 1.0f / lrow[v];
            size_t rowoff = ((size_t)b * Ec + q) * Hh + h * DHh;
            attnOut[rowoff + l16] = (_Float16)(o0[v] * inv);
            attnOut[rowoff + 16 + l16] = (_Float16)(o1[v] * inv);
        }
    }
}

// ---------------------------------------------------------------------------
// LayerNorm over H=256: one wave per row
__global__ __launch_bounds__(256)
void ln_kernel(const float* __restrict__ in, const float* __restrict__ g,
               const float* __restrict__ bta, float* __restrict__ outF32,
               _Float16* __restrict__ outF16) {
    int tid = threadIdx.x, lane = tid & 31, w = tid >> 5;
    int row = blockIdx.x * 8 + w;
    const float* x = in + (size_t)row * Hh;
    float vals[8];
    float s = 0.f, s2 = 0.f;
    #pragma unroll
    for (int i = 0; i < 8; ++i) {
        float t = x[lane + i * 32];
        vals[i] = t; s += t; s2 += t * t;
    }
    #pragma unroll
    for (int m = 16; m >= 1; m >>= 1) { s += __shfl_xor(s, m, 32); s2 += __shfl_xor(s2, m, 32); }
    float mean = s * (1.0f / Hh);
    float var = s2 * (1.0f / Hh) - mean * mean;
    float r = rsqrtf(var + 1e-5f);
    #pragma unroll
    for (int i = 0; i < 8; ++i) {
        int c = lane + i * 32;
        float y = (vals[i] - mean) * r * g[c] + bta[c];
        outF32[(size_t)row * Hh + c] = y;
        outF16[(size_t)row * Hh + c] = (_Float16)y;
    }
}

// ---------------------------------------------------------------------------
// classifier head 512->4 + symmetric scatter into logits [B,V,V,4]
__global__ void cls2_kernel(const _Float16* __restrict__ h1, const float* __restrict__ w2,
                            const float* __restrict__ b2, float* __restrict__ out) {
    int idx = blockIdx.x * blockDim.x + threadIdx.x;  // Mrows*4
    if (idx >= Mrows * 4) return;
    int be = idx >> 2, c = idx & 3;
    int b = be / Ec, e = be - b * Ec;
    float acc = b2[c];
    const _Float16* hp = h1 + (size_t)be * EHh;
    for (int q = 0; q < EHh; ++q) acc += (float)hp[q] * w2[c * EHh + q];
    int i = 0, rem = e;
    while (rem >= Vv - 1 - i) { rem -= Vv - 1 - i; ++i; }
    int j = i + 1 + rem;
    out[(((size_t)b * Vv + i) * Vv + j) * 4 + c] = acc;
    out[(((size_t)b * Vv + j) * Vv + i) * 4 + c] = acc;
}

// ---------------------------------------------------------------------------
extern "C" void kernel_launch(void* const* d_in, const int* in_sizes, int n_in,
                              void* d_out, int out_size, void* d_ws, size_t ws_size,
                              hipStream_t stream) {
    (void)in_sizes; (void)n_in; (void)out_size; (void)ws_size;
    const float* base_adj = (const float*)d_in[0];
    const int*   obs      = (const int*)d_in[2];
    const float* t        = (const float*)d_in[3];
    const float* tok_emb  = (const float*)d_in[4];
    const float* idx_i    = (const float*)d_in[5];
    const float* idx_j    = (const float*)d_in[6];
    const float* deg_w1   = (const float*)d_in[7];
    const float* deg_b1   = (const float*)d_in[8];
    const float* deg_w2   = (const float*)d_in[9];
    const float* deg_b2   = (const float*)d_in[10];
    const float* time_w1  = (const float*)d_in[11];
    const float* time_b1  = (const float*)d_in[12];
    const float* time_w2  = (const float*)d_in[13];
    const float* time_b2  = (const float*)d_in[14];
    const float* qkv_w    = (const float*)d_in[15];
    const float* qkv_b    = (const float*)d_in[16];
    const float* out_w    = (const float*)d_in[17];
    const float* out_b    = (const float*)d_in[18];
    const float* ff1_w    = (const float*)d_in[19];
    const float* ff1_b    = (const float*)d_in[20];
    const float* ff2_w    = (const float*)d_in[21];
    const float* ff2_b    = (const float*)d_in[22];
    const float* ln1_g    = (const float*)d_in[23];
    const float* ln1_b    = (const float*)d_in[24];
    const float* ln2_g    = (const float*)d_in[25];
    const float* ln2_b    = (const float*)d_in[26];
    const float* cls_w1   = (const float*)d_in[27];
    const float* cls_b1   = (const float*)d_in[28];
    const float* cls_w2   = (const float*)d_in[29];
    const float* cls_b2   = (const float*)d_in[30];
    float* out = (float*)d_out;

    char* ws = (char*)d_ws;
    size_t off = 0;
    auto alloc = [&](size_t bytes) -> char* {
        char* p = ws + off;
        off += (bytes + 255) & ~(size_t)255;
        return p;
    };
    float*    xf32  = (float*)   alloc((size_t)Mrows * Hh * 4);
    _Float16* xf16  = (_Float16*)alloc((size_t)Mrows * Hh * 2);
    float*    tmp   = (float*)   alloc((size_t)Mrows * Hh * 4);
    _Float16* Qbuf  = (_Float16*)alloc((size_t)Bb * NHh * Ec * DHh * 2);
    _Float16* Kbuf  = (_Float16*)alloc((size_t)Bb * NHh * Ec * DHh * 2);
    _Float16* Vbuf  = (_Float16*)alloc((size_t)Bb * NHh * Ec * DHh * 2);
    _Float16* attnb = (_Float16*)alloc((size_t)Mrows * Hh * 2);
    _Float16* hbig  = (_Float16*)alloc((size_t)Mrows * 4 * Hh * 2);
    float*    tvec  = (float*)   alloc((size_t)Bb * Hh * 4);
    float*    thid  = (float*)   alloc((size_t)Bb * Hh * 4);
    float*    dnorm = (float*)   alloc((size_t)Bb * Vv * 4);
    _Float16* wqkv  = (_Float16*)alloc((size_t)Ll * 3 * Hh * Hh * 2);
    _Float16* wout  = (_Float16*)alloc((size_t)Ll * Hh * Hh * 2);
    _Float16* wff1  = (_Float16*)alloc((size_t)Ll * 4 * Hh * Hh * 2);
    _Float16* wff2  = (_Float16*)alloc((size_t)Ll * Hh * 4 * Hh * 2);
    _Float16* wcls1 = (_Float16*)alloc((size_t)EHh * Hh * 2);

    // zero logits (diagonal stays 0)
    {
        int n = Bb * Vv * Vv * 4;
        fill_zero_kernel<<<(n + 255) / 256, 256, 0, stream>>>(out, n);
    }
    // weight conversions (layout preserved: [out][in] == GEMM's [N][K])
    {
        int n;
        n = Ll * 3 * Hh * Hh;
        wt_convert_kernel<<<(n + 255) / 256, 256, 0, stream>>>(qkv_w, wqkv, n);
        n = Ll * Hh * Hh;
        wt_convert_kernel<<<(n + 255) / 256, 256, 0, stream>>>(out_w, wout, n);
        n = Ll * 4 * Hh * Hh;
        wt_convert_kernel<<<(n + 255) / 256, 256, 0, stream>>>(ff1_w, wff1, n);
        n = Ll * Hh * 4 * Hh;
        wt_convert_kernel<<<(n + 255) / 256, 256, 0, stream>>>(ff2_w, wff2, n);
        n = EHh * Hh;
        wt_convert_kernel<<<(n + 255) / 256, 256, 0, stream>>>(cls_w1, wcls1, n);
    }

    // scalar embeddings
    time_hidden_kernel<<<Bb, 256, 0, stream>>>(t, time_w1, time_b1, thid);
    time_vec_kernel<<<Bb, 256, 0, stream>>>(thid, time_w2, time_b2, tvec);
    deg_kernel<<<Bb, Vv, 0, stream>>>(base_adj, dnorm);
    embed_kernel<<<Mrows, 256, 0, stream>>>(obs, tok_emb, idx_i, idx_j, dnorm, tvec,
                                            deg_w1, deg_b1, deg_w2, deg_b2, xf32, xf16);

    dim3 blk(256);
    for (int l = 0; l < Ll; ++l) {
        // QKV projection -> scattered per-head f16 Q/K/V
        gemm_kernel<<<dim3(3 * Hh / 64, Mrows / 128), blk, 0, stream>>>(
            xf16, wqkv + (size_t)l * 3 * Hh * Hh, qkv_b + (size_t)l * 3 * Hh,
            Mrows, 3 * Hh, Hh, /*mode*/2, /*act*/0,
            nullptr, nullptr, nullptr, Qbuf, Kbuf, Vbuf);
        // attention
        attn_kernel<<<dim3((Ec + 127) / 128, Bb * NHh), blk, 0, stream>>>(Qbuf, Kbuf, Vbuf, attnb);
        // out projection + residual
        gemm_kernel<<<dim3(Hh / 64, Mrows / 128), blk, 0, stream>>>(
            attnb, wout + (size_t)l * Hh * Hh, out_b + (size_t)l * Hh,
            Mrows, Hh, Hh, /*mode*/0, /*act*/0,
            tmp, xf32, nullptr, nullptr, nullptr, nullptr);
        ln_kernel<<<Mrows / 8, blk, 0, stream>>>(tmp, ln1_g + (size_t)l * Hh, ln1_b + (size_t)l * Hh,
                                                 xf32, xf16);
        // FFN
        gemm_kernel<<<dim3(4 * Hh / 64, Mrows / 128), blk, 0, stream>>>(
            xf16, wff1 + (size_t)l * 4 * Hh * Hh, ff1_b + (size_t)l * 4 * Hh,
            Mrows, 4 * Hh, Hh, /*mode*/1, /*act*/1,
            nullptr, nullptr, hbig, nullptr, nullptr, nullptr);
        gemm_kernel<<<dim3(Hh / 64, Mrows / 128), blk, 0, stream>>>(
            hbig, wff2 + (size_t)l * Hh * 4 * Hh, ff2_b + (size_t)l * Hh,
            Mrows, Hh, 4 * Hh, /*mode*/0, /*act*/0,
            tmp, xf32, nullptr, nullptr, nullptr, nullptr);
        ln_kernel<<<Mrows / 8, blk, 0, stream>>>(tmp, ln2_g + (size_t)l * Hh, ln2_b + (size_t)l * Hh,
                                                 xf32, xf16);
    }
    // classifier: 256 -> 512 (gelu) via WMMA, then 512 -> 4 + symmetric scatter
    gemm_kernel<<<dim3(EHh / 64, Mrows / 128), blk, 0, stream>>>(
        xf16, wcls1, cls_b1, Mrows, EHh, Hh, /*mode*/1, /*act*/1,
        nullptr, nullptr, hbig, nullptr, nullptr, nullptr);
    cls2_kernel<<<(Mrows * 4 + 255) / 256, 256, 0, stream>>>(hbig, cls_w2, cls_b2, out);
}
